// QuantumDMRGLayer_54838142435804
// MI455X (gfx1250) — matrix operand order repaired and checked by reading, not compiled, not run on hardware
//
#include <hip/hip_runtime.h>
#include <hip/hip_bf16.h>
#include <stdint.h>

typedef __attribute__((ext_vector_type(16))) _Float16 v16h;
typedef __attribute__((ext_vector_type(2)))  __fp16   v2fp16;
typedef __attribute__((ext_vector_type(8)))  float    v8f;

#define DIMVEC   256
#define POSL     128
#define NBL      10
#define MB       32
#define BATCH    4096
#define NLSTEPS  127               // W_left transfer steps (sites 1..127)
#define NRSTEPS  126               // W_right transfer steps (sites 129..254)

// Workspace layout (dwords, each dword = packed f16x2), A-operand order:
//   [step][p(2)][jtile(2)][lane(32)][vgpr(8)]
#define TILE_DW  (2*2*32*8)        // 1024 dwords per step
#define LEFT_DW  (NLSTEPS*TILE_DW)
#define RIGHT_DW (NRSTEPS*TILE_DW)

// Single-instruction f32x2 -> packed f16x2 (v_cvt_pk_rtz_f16_f32)
static __device__ __forceinline__ uint32_t pack_f16x2(float a, float b) {
  union { v2fp16 h; uint32_t u; } c;
  c.h = __builtin_amdgcn_cvt_pkrtz(a, b);
  return c.u;
}

// ---------------------------------------------------------------------------
// Prep: swizzle fp32 weights into the WMMA A-operand f16 layout.
// A-matrix 16x32 f16 order used here (contiguous k, not ISA table order): lane = m + 16*h (m=row 0..15),
// VGPR v holds K pair K0,K0+1 with K0 = 16*h + 2*v (same k order as the B operand).
// Left sweep uses A = W^T (A[m][k] = W[k][m]); right sweep uses A = W.
// ---------------------------------------------------------------------------
__global__ void mps_prep_weights(const float* __restrict__ Wl,
                                 const float* __restrict__ Wr,
                                 uint32_t* __restrict__ ws) {
  int d = blockIdx.x * blockDim.x + threadIdx.x;
  int total = LEFT_DW + RIGHT_DW;
  if (d >= total) return;
  bool left = d < LEFT_DW;
  int dd   = left ? d : d - LEFT_DW;
  int v    = dd & 7;
  int lane = (dd >> 3) & 31;
  int T    = (dd >> 8) & 1;
  int p    = (dd >> 9) & 1;
  int t    = dd >> 10;
  int m    = T * 16 + (lane & 15);
  int h    = lane >> 4;
  int K0   = 16 * h + 2 * v;
  float a0, a1;
  if (left) {
    const float* W = Wl + (size_t)((t * 2 + p) * 32) * 32;  // [i][j]
    a0 = W[(size_t)K0 * 32 + m];           // A = W^T
    a1 = W[(size_t)(K0 + 1) * 32 + m];
  } else {
    const float* W = Wr + (size_t)((t * 2 + p) * 32) * 32;
    a0 = W[(size_t)m * 32 + K0];           // A = W
    a1 = W[(size_t)m * 32 + K0 + 1];
  }
  ws[d] = pack_f16x2(a0, a1);
}

// ---------------------------------------------------------------------------
// Main: 64 WGs x 256 threads. Waves 0-3: left sweep (16 rows each);
// waves 4-7: right sweep (same rows). State kept transposed (bond x batch)
// as the WMMA B operand; D accumulator has batch in lanes so per-sample
// scales are plain per-lane packed FMAs.
// ---------------------------------------------------------------------------
__global__ __launch_bounds__(256) void mps_forward(
    const float* __restrict__ x,        // [B][256][2]
    const float* __restrict__ Wfirst,   // [2][32]
    const float* __restrict__ Wlabel,   // [2][32][32][10]
    const float* __restrict__ Wlast,    // [2][32]
    const uint32_t* __restrict__ wsA,
    float* __restrict__ out)            // [B][10]
{
  __shared__ float ldsV[64][MB];
  __shared__ float ldsU[64][MB];

  const int tid     = threadIdx.x;
  const int lane    = tid & 31;
  const int wave    = tid >> 5;
  const bool isRight = wave >= 4;
  const int bloc    = (wave & 3) * 16 + (lane & 15);   // local row 0..63
  const int b       = blockIdx.x * 64 + bloc;          // global batch row
  const float* xb   = x + (size_t)b * (DIMVEC * 2);

  // ---- boundary vector in B-operand layout (8 packed f16x2) ----
  // B 32x16 f16 order used here (contiguous k): VGPR v, lanes0-15: K=2v,2v+1, N=lane;
  //                               lanes16-31: K=16+2v,16+2v+1, N=lane-16.
  union BU { uint32_t u[8]; v16h h; } Bv;
  {
    const float* Wb = isRight ? Wlast : Wfirst;        // [p][i]
    const int t0 = isRight ? (DIMVEC - 1) : 0;
    float x0 = xb[t0 * 2 + 0], x1 = xb[t0 * 2 + 1];
#pragma unroll
    for (int v = 0; v < 8; ++v) {
      int i0 = 2 * v + ((lane < 16) ? 0 : 16);
      float e0 = x0 * Wb[i0]     + x1 * Wb[32 + i0];
      float e1 = x0 * Wb[i0 + 1] + x1 * Wb[32 + i0 + 1];
      Bv.u[v] = pack_f16x2(e0, e1);
    }
  }

  const int nsteps        = isRight ? NRSTEPS : NLSTEPS;
  const uint32_t* wbase   = wsA + (isRight ? LEFT_DW : 0);
  float vn[16];                                        // combined f32 state

  for (int s = 0; s < nsteps; ++s) {
    const int t  = isRight ? (DIMVEC - 2 - s) : (1 + s);
    const int tw = isRight ? (t - (POSL + 1)) : (t - 1);
    const float2 xs = *(const float2*)(xb + t * 2);    // per-sample scales
    const uint32_t* wstep = wbase + (size_t)tw * TILE_DW;

    v8f D[2][2];
#pragma unroll
    for (int p = 0; p < 2; ++p)
#pragma unroll
      for (int T = 0; T < 2; ++T) {
        union BU A;
        const uint4* ap = (const uint4*)(wstep + ((p * 2 + T) * 32 + lane) * 8);
        *(uint4*)&A.u[0] = ap[0];
        *(uint4*)&A.u[4] = ap[1];
        v8f c = {};
        D[p][T] = __builtin_amdgcn_wmma_f32_16x16x32_f16(
            false, A.h, false, Bv.h, (short)0, c, false, false);
      }

    // v' = x0*D0 + x1*D1  (D: j in vgpr index / lane half, batch in lane%16)
#pragma unroll
    for (int r = 0; r < 8; ++r) {
      vn[r]     = xs.x * D[0][0][r] + xs.y * D[1][0][r];  // j = r / r+8
      vn[8 + r] = xs.x * D[0][1][r] + xs.y * D[1][1][r];  // j = 16+r / 24+r
    }

    // repack D layout -> next B operand: pack f16 pairs, swap lane halves
    uint32_t q[8];
#pragma unroll
    for (int v = 0; v < 8; ++v)
      q[v] = pack_f16x2(vn[2 * v], vn[2 * v + 1]);
#pragma unroll
    for (int v = 0; v < 4; ++v) {
      // ds_swizzle SWAPX16: xor=0x10, and=0x1f -> offset 0x401F
      uint32_t sa = (uint32_t)__builtin_amdgcn_ds_swizzle((int)q[v],     0x401f);
      uint32_t sb = (uint32_t)__builtin_amdgcn_ds_swizzle((int)q[v + 4], 0x401f);
      Bv.u[v]     = (lane < 16) ? q[v] : sb;  // lo: K 2v,2v+1   hi: K 16+2v
      Bv.u[v + 4] = (lane < 16) ? sa   : q[v + 4];
    }
  }

  // ---- store final boundary vectors (fp32) to LDS ----
  {
    float (*ldsT)[MB] = isRight ? ldsU : ldsV;
#pragma unroll
    for (int r = 0; r < 16; ++r) {
      int j = ((r < 8) ? r : (r + 8)) + ((lane < 16) ? 0 : 8);
      ldsT[bloc][j] = vn[r];
    }
  }
  __syncthreads();

  // ---- label-site contraction: out[b,l] = x_p vL_i uR_j Wlabel[p,i,j,l] ----
  for (int idx = tid; idx < 64 * NBL; idx += 256) {
    int bl = idx / NBL, l = idx % NBL;
    int bg = blockIdx.x * 64 + bl;
    const float* xl = x + (size_t)bg * (DIMVEC * 2) + POSL * 2;
    float x0 = xl[0], x1 = xl[1];
    float acc = 0.f;
    for (int i = 0; i < MB; ++i) {
      float vi = ldsV[bl][i];
      const float* w0 = Wlabel + ((size_t)(0 * 32 + i) * 32) * NBL + l;
      const float* w1 = Wlabel + ((size_t)(32 + i) * 32) * NBL + l;
      float inner = 0.f;
      for (int j = 0; j < MB; ++j)
        inner += ldsU[bl][j] * (x0 * w0[(size_t)j * NBL] + x1 * w1[(size_t)j * NBL]);
      acc += vi * inner;
    }
    out[(size_t)bg * NBL + l] = acc;
  }
}

// ---------------------------------------------------------------------------
extern "C" void kernel_launch(void* const* d_in, const int* in_sizes, int n_in,
                              void* d_out, int out_size, void* d_ws, size_t ws_size,
                              hipStream_t stream) {
  (void)in_sizes; (void)n_in; (void)out_size; (void)ws_size;
  const float* x     = (const float*)d_in[0];   // (4096,256,2)
  const float* Wf    = (const float*)d_in[1];   // (2,32)
  const float* Wl    = (const float*)d_in[2];   // (127,2,32,32)
  const float* Wlab  = (const float*)d_in[3];   // (2,32,32,10)
  const float* Wr    = (const float*)d_in[4];   // (126,2,32,32)
  const float* Wlast = (const float*)d_in[5];   // (2,32)
  uint32_t* ws = (uint32_t*)d_ws;               // needs ~1.01 MB

  int total = LEFT_DW + RIGHT_DW;
  mps_prep_weights<<<(total + 255) / 256, 256, 0, stream>>>(Wl, Wr, ws);
  mps_forward<<<BATCH / 64, 256, 0, stream>>>(x, Wf, Wlab, Wlast, ws, (float*)d_out);
}
